// GAT_59768764891271
// MI455X (gfx1250) — matrix operand order repaired and checked by reading, not compiled, not run on hardware
//
#include <hip/hip_runtime.h>
#include <hip/hip_bf16.h>
#include <math.h>

#define N_NODES 50000
#define N_EDGES 800000
#define IN_DIM  256
#define HEADS   4
#define HID     64
#define OUT_DIM (HEADS * HID)   // 256
#define NEG_SLOPE 0.2f

typedef __attribute__((ext_vector_type(16))) _Float16 v16h;
typedef __attribute__((ext_vector_type(8)))  _Float16 v8h;
typedef __attribute__((ext_vector_type(8)))  float    v8f;

// ---- order-preserving float <-> uint encoding (for deterministic atomic max) ----
__device__ __forceinline__ unsigned enc_f32(float v) {
    unsigned u = __float_as_uint(v);
    return (u & 0x80000000u) ? ~u : (u | 0x80000000u);
}
__device__ __forceinline__ float dec_f32(unsigned e) {
    return (e & 0x80000000u) ? __uint_as_float(e & 0x7FFFFFFFu)
                             : __uint_as_float(~e);
}
#define ENC_NEG_INF 0x007FFFFFu   // enc_f32(-inf)

// ============================================================================
// K0: initialize out (N*64), denom (N*4) to 0 and m_enc (N*4) to enc(-inf)
// ============================================================================
__global__ void gat_init(float* __restrict__ out, float* __restrict__ denom,
                         unsigned* __restrict__ m_enc) {
    int i = blockIdx.x * blockDim.x + threadIdx.x;
    if (i < N_NODES * HID) out[i] = 0.0f;
    if (i < N_NODES * HEADS) { denom[i] = 0.0f; m_enc[i] = ENC_NEG_INF; }
}

// ============================================================================
// K0b: one-shot f32 -> f16 conversion.
//   xh  = f16 copy of x                  [N_NODES, IN_DIM]
//   Wth = f16 TRANSPOSED copy of W       [OUT_DIM, IN_DIM]  (row = output col)
// Makes every WMMA fragment a contiguous 128-bit load.
// ============================================================================
__global__ void gat_cvt_f16(const float* __restrict__ x,
                            const float* __restrict__ W,
                            _Float16* __restrict__ xh,
                            _Float16* __restrict__ Wth) {
    int i = blockIdx.x * blockDim.x + threadIdx.x;
    if (i < N_NODES * IN_DIM) xh[i] = (_Float16)x[i];
    if (i < IN_DIM * OUT_DIM) {
        int n = i / IN_DIM, k = i - n * IN_DIM;
        Wth[i] = (_Float16)W[(size_t)k * OUT_DIM + n];   // Wth[n][k] = W[k][n]
    }
}

// ============================================================================
// K1: feat = x @ W via v_wmma_f32_16x16x32_f16.
// One wave per 16-row M-stripe; computes ALL 16 N-tiles (16 v8f accumulators)
// so the A fragment is loaded once per k-step and reused by 16 WMMAs.
//   A frag (16x32 f16): lane l (group g) holds K = k0+g*16+{0..7} (VGPR0-3)
//                       and K = k0+g*16+8+{0..7} (VGPR4-7)  -> two b128 loads
//   B frag (32x16 f16): lane l = col, K = k0+g*16+{0..15}    -> one 32B load
//   C/D: VGPR r -> M = r + 8*g, N = l
// ============================================================================
__global__ void __launch_bounds__(256)
gat_gemm_wmma(const _Float16* __restrict__ xh,
              const _Float16* __restrict__ Wth,
              float* __restrict__ feat) {
    const int waveId = blockIdx.x * (blockDim.x >> 5) + (threadIdx.x >> 5);
    if (waveId >= N_NODES / 16) return;
    const int tileM = waveId * 16;
    const int lane  = threadIdx.x & 31;
    const int g = lane >> 4;
    const int l = lane & 15;

    v8f acc[16];
#pragma unroll
    for (int nt = 0; nt < 16; ++nt) acc[nt] = (v8f){};

    const _Float16* __restrict__ arow = xh + (size_t)(tileM + l) * IN_DIM;

    for (int k0 = 0; k0 < IN_DIM; k0 += 32) {
        v8h alo = *(const v8h*)(arow + k0 + g * 16);
        v8h ahi = *(const v8h*)(arow + k0 + g * 16 + 8);
        v16h a  = __builtin_shufflevector(alo, ahi,
                      0, 1, 2, 3, 4, 5, 6, 7, 8, 9, 10, 11, 12, 13, 14, 15);
#pragma unroll
        for (int nt = 0; nt < 16; ++nt) {
            v16h b = *(const v16h*)(Wth + (size_t)(nt * 16 + l) * IN_DIM
                                        + k0 + g * 16);
            acc[nt] = __builtin_amdgcn_wmma_f32_16x16x32_f16(
                /*neg_a=*/false, a, /*neg_b=*/false, b,
                /*c_mod=*/(short)0, acc[nt],
                /*reuse_a=*/false, /*reuse_b=*/false);
        }
    }
#pragma unroll
    for (int nt = 0; nt < 16; ++nt)
#pragma unroll
        for (int r = 0; r < 8; ++r)
            feat[(size_t)(tileM + r + 8 * g) * OUT_DIM + nt * 16 + l] = acc[nt][r];
}

// ============================================================================
// K2: per-node attention logits  el[n,h] = feat[n,h,:] . attn_l[h,:]
// ============================================================================
__global__ void gat_logits(const float* __restrict__ feat,
                           const float* __restrict__ al,
                           const float* __restrict__ ar,
                           float* __restrict__ el, float* __restrict__ er) {
    int i = blockIdx.x * blockDim.x + threadIdx.x;   // i = n*HEADS + h
    if (i >= N_NODES * HEADS) return;
    int h = i & (HEADS - 1);
    const float* __restrict__ f = feat + (size_t)(i >> 2) * OUT_DIM + h * HID;
    float sl = 0.0f, sr = 0.0f;
#pragma unroll 8
    for (int d = 0; d < HID; ++d) {
        float v = f[d];
        sl = fmaf(v, al[h * HID + d], sl);
        sr = fmaf(v, ar[h * HID + d], sr);
    }
    el[i] = sl;
    er[i] = sr;
}

// ============================================================================
// K3: edge scores e = leaky_relu(el[src]+er[dst]); atomic segment-max into m_enc
// ============================================================================
__global__ void gat_edge_score(const float* __restrict__ el,
                               const float* __restrict__ er,
                               const int* __restrict__ src,
                               const int* __restrict__ dst,
                               float* __restrict__ eval,
                               unsigned* __restrict__ m_enc) {
    int e = blockIdx.x * blockDim.x + threadIdx.x;
    if (e >= N_EDGES) return;
    int s = src[e], d = dst[e];
#pragma unroll
    for (int h = 0; h < HEADS; ++h) {
        float v = el[s * HEADS + h] + er[d * HEADS + h];
        v = (v > 0.0f) ? v : NEG_SLOPE * v;
        eval[(size_t)e * HEADS + h] = v;
        atomicMax(&m_enc[d * HEADS + h], enc_f32(v));
    }
}

// ============================================================================
// K4: decode segment max; zero-in-degree nodes (still -inf) -> 0  (matches ref)
// ============================================================================
__global__ void gat_decode_m(const unsigned* __restrict__ m_enc,
                             float* __restrict__ m) {
    int i = blockIdx.x * blockDim.x + threadIdx.x;
    if (i >= N_NODES * HEADS) return;
    unsigned e = m_enc[i];
    m[i] = (e == ENC_NEG_INF) ? 0.0f : dec_f32(e);
}

// ============================================================================
// K5: ex = exp(e - m[dst]) in place; atomic segment-sum into denom
// ============================================================================
__global__ void gat_edge_exp(const float* __restrict__ m,
                             const int* __restrict__ dst,
                             float* __restrict__ eval,     // in: e, out: exp
                             float* __restrict__ denom) {
    int e = blockIdx.x * blockDim.x + threadIdx.x;
    if (e >= N_EDGES) return;
    int d = dst[e];
#pragma unroll
    for (int h = 0; h < HEADS; ++h) {
        float ex = expf(eval[(size_t)e * HEADS + h] - m[d * HEADS + h]);
        eval[(size_t)e * HEADS + h] = ex;
        atomicAdd(&denom[d * HEADS + h], ex);
    }
}

// ============================================================================
// K6: aggregation.  64 threads per edge (one per d); sum over heads with the
//     1/HEADS mean folded in, single f32 atomic per (edge, d).
//     feat (51.2MB) and out (12.8MB) are L2-resident on MI455X (192MB L2).
// ============================================================================
__global__ void gat_aggregate(const float* __restrict__ feat,
                              const float* __restrict__ eval,   // exp values
                              const float* __restrict__ denom,
                              const int* __restrict__ src,
                              const int* __restrict__ dst,
                              float* __restrict__ out) {
    long long idx = (long long)blockIdx.x * blockDim.x + threadIdx.x;
    int e  = (int)(idx >> 6);
    int dd = (int)(idx & 63);
    if (e >= N_EDGES) return;
    int s = src[e], d = dst[e];
    float acc = 0.0f;
#pragma unroll
    for (int h = 0; h < HEADS; ++h) {
        float alpha = eval[(size_t)e * HEADS + h] / denom[d * HEADS + h];
        acc = fmaf(alpha, feat[(size_t)s * OUT_DIM + h * HID + dd], acc);
    }
    atomicAdd(&out[(size_t)d * HID + dd], acc * (1.0f / HEADS));
}

// ============================================================================
extern "C" void kernel_launch(void* const* d_in, const int* in_sizes, int n_in,
                              void* d_out, int out_size, void* d_ws, size_t ws_size,
                              hipStream_t stream) {
    const float* x   = (const float*)d_in[0];
    const float* W   = (const float*)d_in[1];
    const float* al  = (const float*)d_in[2];
    const float* ar  = (const float*)d_in[3];
    const int*   src = (const int*)d_in[4];
    const int*   dst = (const int*)d_in[5];
    float*       out = (float*)d_out;      // [N_NODES, HID]

    // --- carve workspace (256B aligned) ---
    uintptr_t p = ((uintptr_t)d_ws + 255) & ~(uintptr_t)255;
    auto carve = [&](size_t bytes) {
        void* r = (void*)p;
        p = (p + bytes + 255) & ~(uintptr_t)255;
        return r;
    };
    float*     feat  = (float*)    carve((size_t)N_NODES * OUT_DIM * 4); // 51.2 MB
    _Float16*  xh    = (_Float16*) carve((size_t)N_NODES * IN_DIM * 2);  // 25.6 MB
    _Float16*  Wth   = (_Float16*) carve((size_t)IN_DIM * OUT_DIM * 2);  // 128 KB
    float*     el    = (float*)    carve((size_t)N_NODES * HEADS * 4);
    float*     er    = (float*)    carve((size_t)N_NODES * HEADS * 4);
    unsigned*  m_enc = (unsigned*) carve((size_t)N_NODES * HEADS * 4);
    float*     m     = (float*)    carve((size_t)N_NODES * HEADS * 4);
    float*     denom = (float*)    carve((size_t)N_NODES * HEADS * 4);
    float*     eval  = (float*)    carve((size_t)N_EDGES * HEADS * 4);   // 12.8 MB

    const int B = 256;

    // K0: init accumulators
    {
        int n = N_NODES * HID;
        gat_init<<<(n + B - 1) / B, B, 0, stream>>>(out, denom, m_enc);
    }
    // K0b: f32 -> f16 conversions (x, and W transposed)
    {
        int n = N_NODES * IN_DIM;
        gat_cvt_f16<<<(n + B - 1) / B, B, 0, stream>>>(x, W, xh, Wth);
    }
    // K1: WMMA GEMM  (3125 M-stripe waves, 8 waves/block)
    {
        int waves  = N_NODES / 16;                 // 3125
        int blocks = (waves + (B / 32) - 1) / (B / 32);
        gat_gemm_wmma<<<blocks, B, 0, stream>>>(xh, Wth, feat);
    }
    // K2: node logits
    {
        int n = N_NODES * HEADS;
        gat_logits<<<(n + B - 1) / B, B, 0, stream>>>(feat, al, ar, el, er);
    }
    // K3: edge scores + segment max
    gat_edge_score<<<(N_EDGES + B - 1) / B, B, 0, stream>>>(el, er, src, dst, eval, m_enc);
    // K4: decode max
    {
        int n = N_NODES * HEADS;
        gat_decode_m<<<(n + B - 1) / B, B, 0, stream>>>(m_enc, m);
    }
    // K5: exp + segment sum
    gat_edge_exp<<<(N_EDGES + B - 1) / B, B, 0, stream>>>(m, dst, eval, denom);
    // K6: weighted scatter-aggregate with head-mean folded in
    {
        long long thr = (long long)N_EDGES * HID;
        int blocks = (int)((thr + B - 1) / B);
        gat_aggregate<<<blocks, B, 0, stream>>>(feat, eval, denom, src, dst, out);
    }
}
